// GATLayer_23768349016159
// MI455X (gfx1250) — matrix operand order repaired and checked by reading, not compiled, not run on hardware
//
#include <hip/hip_runtime.h>
#include <cstdint>
#include <cstddef>

#define N_NODES 12288
#define F_IN    256
#define F_OUT   64
#define LEAKY   0.01f

typedef __attribute__((ext_vector_type(16))) __bf16 v16bf;
typedef __attribute__((ext_vector_type(8)))  float  v8f;

union BfPack {
    v16bf  v;
    __bf16 h[16];
    uint4  q[2];
};

__device__ __forceinline__ float lrelu(float x) { return x > 0.f ? x : LEAKY * x; }

// ---------------------------------------------------------------------------
// Kernel 1: Wh = h @ w via WMMA bf16. Emits WhT (bf16, [64][12288]) so the
// big pass reads B-tiles as contiguous 32B chunks, plus Wh1 = Wh@a[:64],
// Wh2 = Wh@a[64:] via in-register cross-lane reductions.
// One wave handles 16 rows x 64 cols, K=256 in 8 steps.
// ---------------------------------------------------------------------------
__global__ __launch_bounds__(32) void k1_wh(const float* __restrict__ h,
                                            const float* __restrict__ w,
                                            const float* __restrict__ avec,
                                            __bf16* __restrict__ whT,
                                            float* __restrict__ wh1,
                                            float* __restrict__ wh2) {
    __shared__ __bf16 wT[F_OUT][F_IN + 16];           // w transposed, bf16, padded
    const int lane = threadIdx.x;

    for (int idx = lane; idx < F_IN * F_OUT; idx += 32) {
        int k = idx >> 6, col = idx & 63;
        wT[col][k] = (__bf16)w[idx];
    }
    __syncthreads();

    const int rowBase = blockIdx.x * 16;
    const int m    = lane & 15;
    const int half = lane >> 4;
    const float* hrow = h + (size_t)(rowBase + m) * F_IN;

    v8f acc[4] = {};

    for (int kb = 0; kb < F_IN; kb += 32) {
        // A tile 16x32: lanes 0-15 hold K 0-7 & 16-23; lanes 16-31 K 8-15 & 24-31.
        BfPack A;
        const float4* r1 = (const float4*)(hrow + kb + 8 * half);
        const float4* r2 = (const float4*)(hrow + kb + 16 + 8 * half);
        float4 f0 = r1[0], f1 = r1[1], f2 = r2[0], f3 = r2[1];
        A.h[0]  = (__bf16)f0.x; A.h[1]  = (__bf16)f0.y;
        A.h[2]  = (__bf16)f0.z; A.h[3]  = (__bf16)f0.w;
        A.h[4]  = (__bf16)f1.x; A.h[5]  = (__bf16)f1.y;
        A.h[6]  = (__bf16)f1.z; A.h[7]  = (__bf16)f1.w;
        A.h[8]  = (__bf16)f2.x; A.h[9]  = (__bf16)f2.y;
        A.h[10] = (__bf16)f2.z; A.h[11] = (__bf16)f2.w;
        A.h[12] = (__bf16)f3.x; A.h[13] = (__bf16)f3.y;
        A.h[14] = (__bf16)f3.z; A.h[15] = (__bf16)f3.w;

#pragma unroll
        for (int t = 0; t < 4; ++t) {
            BfPack B;
            const __bf16* bp = &wT[16 * t + m][kb + 8 * half];
            B.q[0] = *(const uint4*)bp;
            B.q[1] = *(const uint4*)(bp + 16);
            acc[t] = __builtin_amdgcn_wmma_f32_16x16x32_bf16(
                false, A.v, false, B.v, (short)0, acc[t], false, false);
        }
    }

    float p1[8] = {}, p2[8] = {};
#pragma unroll
    for (int t = 0; t < 4; ++t) {
        float a0 = avec[16 * t + m];
        float a1 = avec[F_OUT + 16 * t + m];
#pragma unroll
        for (int r = 0; r < 8; ++r) {
            float vv = acc[t][r];                      // row rowBase+r+8*half, col 16t+m
            p1[r] += vv * a0;
            p2[r] += vv * a1;
            int row = rowBase + r + 8 * half;
            whT[(size_t)(16 * t + m) * N_NODES + row] = (__bf16)vv;
        }
    }
#pragma unroll
    for (int r = 0; r < 8; ++r) {
#pragma unroll
        for (int msk = 1; msk < 16; msk <<= 1) {
            p1[r] += __shfl_xor(p1[r], msk, 32);
            p2[r] += __shfl_xor(p2[r], msk, 32);
        }
        if (m == 0) {
            int row = rowBase + r + 8 * half;
            wh1[row] = p1[r];
            wh2[row] = p2[r];
        }
    }
}

// ---------------------------------------------------------------------------
// Kernel 2: global max of Wh2 (48 KB). Softmax shift-invariance lets us use
// c_i = lrelu(Wh1_i + gmax) as the stabilizer, so no adjacency read is needed
// to compute the per-row shift -> the big pass streams adj exactly ONCE.
// ---------------------------------------------------------------------------
__global__ __launch_bounds__(256) void k2_gmax(const float* __restrict__ wh2,
                                               float* __restrict__ gmaxp) {
    __shared__ float red[256];
    const int tid = threadIdx.x;
    float mx = -3.0e38f;
    for (int j = tid; j < N_NODES; j += 256) mx = fmaxf(mx, wh2[j]);
    red[tid] = mx;
    __syncthreads();
    for (int s = 128; s > 0; s >>= 1) {
        if (tid < s) red[tid] = fmaxf(red[tid], red[tid + s]);
        __syncthreads();
    }
    if (tid == 0) gmaxp[0] = red[0];
}

// ---------------------------------------------------------------------------
// Kernel 3 (fused single pass): out = relu( (masked exp-weights) @ Wh / sum ).
// adj streamed from HBM once (604 MB -> ~26 us roofline). Row softmax sums
// accumulated in-register alongside the WMMA GEMM; remapped from A-row layout
// (lane = row) to C-row layout (vgpr = row) via a 16-float LDS broadcast.
// Wh2 staged into LDS via the gfx1250 async global->LDS path when available.
// ---------------------------------------------------------------------------
__device__ __forceinline__ float quad(BfPack& A, int base, int4 av, float4 wv,
                                      float wh1v, float c) {
    float e0 = lrelu(wh1v + wv.x) - c;
    float e1 = lrelu(wh1v + wv.y) - c;
    float e2 = lrelu(wh1v + wv.z) - c;
    float e3 = lrelu(wh1v + wv.w) - c;
    float v0 = (av.x > 0) ? __expf(e0) : 0.f;
    float v1 = (av.y > 0) ? __expf(e1) : 0.f;
    float v2 = (av.z > 0) ? __expf(e2) : 0.f;
    float v3 = (av.w > 0) ? __expf(e3) : 0.f;
    A.h[base + 0] = (__bf16)v0;
    A.h[base + 1] = (__bf16)v1;
    A.h[base + 2] = (__bf16)v2;
    A.h[base + 3] = (__bf16)v3;
    return (v0 + v1) + (v2 + v3);
}

__global__ __launch_bounds__(32) void k3_fused(const int* __restrict__ adj,
                                               const __bf16* __restrict__ whT,
                                               const float* __restrict__ wh1,
                                               const float* __restrict__ wh2,
                                               const float* __restrict__ gmaxp,
                                               float* __restrict__ out) {
    __shared__ float wh2s[N_NODES];                    // 48 KB
    __shared__ float rs[16];
    const int lane = threadIdx.x;
    {
#if __has_builtin(__builtin_amdgcn_global_load_async_to_lds_b128)
        // ASYNCcnt-tracked bulk copy: 16B per lane per op, no VGPR round-trip.
        // Builtin params are true GCC-style vectors (vector_size(16) int), not
        // HIP_vector_type, in AS(1) (global) / AS(3) (LDS).
        typedef int v4i __attribute__((vector_size(16)));
        typedef __attribute__((address_space(1))) v4i* gas_p;
        typedef __attribute__((address_space(3))) v4i* las_p;
        float* wh2_nc = const_cast<float*>(wh2);
        for (int i = lane; i < N_NODES / 4; i += 32) {
            __builtin_amdgcn_global_load_async_to_lds_b128(
                (gas_p)(wh2_nc + 4 * i),
                (las_p)(wh2s + 4 * i),
                0, 0);
        }
#if __has_builtin(__builtin_amdgcn_s_wait_asynccnt)
        __builtin_amdgcn_s_wait_asynccnt(0);
#endif
#else
        const float4* src = (const float4*)wh2;
        float4* dst = (float4*)wh2s;
        for (int i = lane; i < N_NODES / 4; i += 32) dst[i] = src[i];
#endif
    }
    __syncthreads();

    const int rowBase = blockIdx.x * 16;
    const int m    = lane & 15;
    const int half = lane >> 4;
    const int row  = rowBase + m;
    const float wh1v = wh1[row];
    const float c    = lrelu(wh1v + gmaxp[0]);         // row softmax shift
    const int* arow  = adj + (size_t)row * N_NODES;

    v8f acc[4] = {};
    float ssum = 0.f;

    for (int jb = 0; jb < N_NODES; jb += 32) {
        __builtin_prefetch(arow + jb + 1024, 0, 0);    // global_prefetch_b8 on adj
        // A tile 16x32 bf16: lane's row m; K-slots [8h,8h+8) and [16+8h,16+8h+8)
        const int j0 = jb + 8 * half;
        const int j1 = jb + 16 + 8 * half;
        int4 a0 = *(const int4*)(arow + j0);
        int4 a1 = *(const int4*)(arow + j0 + 4);
        int4 a2 = *(const int4*)(arow + j1);
        int4 a3 = *(const int4*)(arow + j1 + 4);
        float4 w0 = *(const float4*)(&wh2s[j0]);
        float4 w1 = *(const float4*)(&wh2s[j0 + 4]);
        float4 w2 = *(const float4*)(&wh2s[j1]);
        float4 w3 = *(const float4*)(&wh2s[j1 + 4]);

        BfPack A;
        ssum += quad(A, 0,  a0, w0, wh1v, c);
        ssum += quad(A, 4,  a1, w1, wh1v, c);
        ssum += quad(A, 8,  a2, w2, wh1v, c);
        ssum += quad(A, 12, a3, w3, wh1v, c);

#pragma unroll
        for (int t = 0; t < 4; ++t) {
            BfPack B;
            const __bf16* bp = whT + (size_t)(16 * t + m) * N_NODES + jb + 8 * half;
            B.q[0] = *(const uint4*)bp;
            B.q[1] = *(const uint4*)(bp + 16);
            acc[t] = __builtin_amdgcn_wmma_f32_16x16x32_bf16(
                false, A.v, false, B.v, (short)0, acc[t], false, false);
        }
    }

    // Combine the two half-wave K-ranges of each row, broadcast row sums
    ssum += __shfl_xor(ssum, 16, 32);
    if (lane < 16) rs[lane] = ssum;
    __syncthreads();

#pragma unroll
    for (int r = 0; r < 8; ++r) {
        float s = rs[r + 8 * half];
        float inv = (s > 0.f) ? (1.0f / s) : 0.f;      // empty row -> zeros
        int orow = rowBase + r + 8 * half;
#pragma unroll
        for (int t = 0; t < 4; ++t) {
            out[(size_t)orow * F_OUT + 16 * t + m] = fmaxf(acc[t][r] * inv, 0.f);
        }
    }
}

// ---------------------------------------------------------------------------
extern "C" void kernel_launch(void* const* d_in, const int* in_sizes, int n_in,
                              void* d_out, int out_size, void* d_ws, size_t ws_size,
                              hipStream_t stream) {
    (void)in_sizes; (void)n_in; (void)out_size; (void)ws_size;

    const float* h   = (const float*)d_in[0];
    const int*   adj = (const int*)d_in[1];
    const float* w   = (const float*)d_in[2];
    const float* a   = (const float*)d_in[3];
    float* out = (float*)d_out;

    char* ws = (char*)d_ws;
    __bf16* whT = (__bf16*)ws;                                   // 64*12288*2 B
    float* wh1  = (float*)(ws + (size_t)F_OUT * N_NODES * 2);
    float* wh2  = wh1 + N_NODES;
    float* gmax = wh2 + N_NODES;

    k1_wh   <<<dim3(N_NODES / 16), dim3(32),  0, stream>>>(h, w, a, whT, wh1, wh2);
    k2_gmax <<<dim3(1),            dim3(256), 0, stream>>>(wh2, gmax);
    k3_fused<<<dim3(N_NODES / 16), dim3(32),  0, stream>>>(adj, whT, wh1, wh2,
                                                           gmax, out);
}
